// Attention_18433999634671
// MI455X (gfx1250) — matrix operand order repaired and checked by reading, not compiled, not run on hardware
//
#include <hip/hip_runtime.h>

// ---------------------------------------------------------------------------
// MI455X (gfx1250) dual-stream attention.
// bf16 WMMA (v_wmma_f32_16x16x32_bf16) everywhere; f32 accumulate.
// B=8, S=1024, H=768, NH=12, DH=64.  M=8192, N=K=768 for all GEMMs.
// ---------------------------------------------------------------------------

typedef __attribute__((ext_vector_type(16))) __bf16 v16bf;
typedef __attribute__((ext_vector_type(8)))  float  v8f;
typedef __attribute__((ext_vector_type(4)))  int    vi4;

union Frag {
    uint4 u4[2];
    v16bf v;
};

static __device__ __forceinline__ unsigned short f2bf(float f) {
    unsigned int u = __float_as_uint(f);
    u += 0x7FFFu + ((u >> 16) & 1u);      // round-to-nearest-even
    return (unsigned short)(u >> 16);
}

static __device__ __forceinline__ v8f wmma_bf16(const Frag& a, const Frag& b, v8f c) {
    return __builtin_amdgcn_wmma_f32_16x16x32_bf16(
        false, a.v, false, b.v, (short)0, c, false, false);
}

// ---- async global -> LDS (gfx1250), guarded with safe fallback -------------
#if defined(__has_builtin)
#if __has_builtin(__builtin_amdgcn_global_load_async_to_lds_b128)
#define HAVE_ASYNC_LDS 1
#endif
#endif

static __device__ __forceinline__ void async_copy16(const unsigned short* g,
                                                    unsigned short* l) {
#if defined(HAVE_ASYNC_LDS)
    // signature: (v4i addrspace(1)*, v4i addrspace(3)*, imm offset, imm cpol)
    __builtin_amdgcn_global_load_async_to_lds_b128(
        (__attribute__((address_space(1))) vi4*)g,
        (__attribute__((address_space(3))) vi4*)l,
        0 /*offset*/, 0 /*cpol*/);
#else
    *(uint4*)l = *(const uint4*)g;
#endif
}

static __device__ __forceinline__ void async_wait0() {
#if defined(HAVE_ASYNC_LDS)
#if __has_builtin(__builtin_amdgcn_s_wait_asynccnt)
    __builtin_amdgcn_s_wait_asynccnt(0);
#else
    asm volatile("s_wait_asynccnt 0x0" ::: "memory");
#endif
#endif
}

#define MDIM 8192
#define NDIM 768
#define KDIM 768

// ---------------------------------------------------------------------------
// Generic GEMM: C[M,N] = A[M,K](f32) @ W[N,K](f32)^T + bias*bscale
// out mode 0: bf16, split heads  [bh][s][64]        (Q / K layout)
// out mode 1: bf16, transposed   [bh][64][s]        (V^T layout for P@V)
// out mode 2: f32,  dense        [m][768]           (final output)
// Block: 256 thr = 8 waves; tile 128x128; wave tile 64x32 (4x2 WMMA); K step 32.
// ---------------------------------------------------------------------------
__global__ __launch_bounds__(256)
void gemm_bf16wmma(const float* __restrict__ A, const float* __restrict__ W,
                   const float* __restrict__ bias,
                   const float* wsA, const float* wsB,
                   void* __restrict__ Cout, int mode)
{
    __shared__ unsigned short As[128 * 40];   // 32 cols + 8 pad (16B-aligned rows)
    __shared__ unsigned short Bs[128 * 40];

    const int t     = threadIdx.x;
    const int lane  = t & 31;
    const int wave  = t >> 5;
    const int hh    = lane >> 4;     // half-wave id
    const int lr    = lane & 15;
    const int waveM = wave & 1;      // 2 waves along M (64 rows each)
    const int waveN = wave >> 1;     // 4 waves along N (32 cols each)
    const int bm    = blockIdx.x * 128;
    const int bn    = blockIdx.y * 128;

    const int srow = t >> 1, sseg = t & 1;   // staging: 128 rows x (2 x 16 cols)

    v8f acc[4][2] = {};

    for (int kk = 0; kk < KDIM; kk += 32) {
        // ---- prefetch next K tile into caches (global_prefetch_b8) ----
        if (kk + 32 < KDIM) {
            __builtin_prefetch(&A[(size_t)(bm + srow) * KDIM + kk + 32 + sseg * 16], 0, 1);
            __builtin_prefetch(&W[(size_t)(bn + srow) * KDIM + kk + 32 + sseg * 16], 0, 1);
        }
        // ---- stage A and W tiles (f32 -> bf16), identical patterns ----
        {
            const float4* srcA = (const float4*)&A[(size_t)(bm + srow) * KDIM + kk + sseg * 16];
            const float4* srcW = (const float4*)&W[(size_t)(bn + srow) * KDIM + kk + sseg * 16];
            float4 a0 = srcA[0], a1 = srcA[1], a2 = srcA[2], a3 = srcA[3];
            float4 w0 = srcW[0], w1 = srcW[1], w2 = srcW[2], w3 = srcW[3];
            union { unsigned short s[16]; uint4 u[2]; } ca, cw;
            float fa[16] = { a0.x,a0.y,a0.z,a0.w, a1.x,a1.y,a1.z,a1.w,
                             a2.x,a2.y,a2.z,a2.w, a3.x,a3.y,a3.z,a3.w };
            float fw[16] = { w0.x,w0.y,w0.z,w0.w, w1.x,w1.y,w1.z,w1.w,
                             w2.x,w2.y,w2.z,w2.w, w3.x,w3.y,w3.z,w3.w };
            #pragma unroll
            for (int i = 0; i < 16; ++i) { ca.s[i] = f2bf(fa[i]); cw.s[i] = f2bf(fw[i]); }
            uint4* dA = (uint4*)&As[srow * 40 + sseg * 16];
            uint4* dW = (uint4*)&Bs[srow * 40 + sseg * 16];
            dA[0] = ca.u[0];  dA[1] = ca.u[1];
            dW[0] = cw.u[0];  dW[1] = cw.u[1];
        }
        __syncthreads();

        // ---- fragments (documented 16-bit A 16x32 / B 32x16 layouts) ----
        Frag af[4], bw[2];
        #pragma unroll
        for (int i = 0; i < 4; ++i) {
            int row = waveM * 64 + i * 16 + lr;
            af[i].u4[0] = *(const uint4*)&As[row * 40 + hh * 8];        // K 0..7 / 8..15
            af[i].u4[1] = *(const uint4*)&As[row * 40 + 16 + hh * 8];   // K 16..23 / 24..31
        }
        #pragma unroll
        for (int j = 0; j < 2; ++j) {
            int col = waveN * 32 + j * 16 + lr;
            bw[j].u4[0] = *(const uint4*)&Bs[col * 40 + hh * 8];        // K 0..7 / 8..15
            bw[j].u4[1] = *(const uint4*)&Bs[col * 40 + 16 + hh * 8];
        }
        #pragma unroll
        for (int i = 0; i < 4; ++i)
            #pragma unroll
            for (int j = 0; j < 2; ++j)
                acc[i][j] = wmma_bf16(af[i], bw[j], acc[i][j]);
        __syncthreads();
    }

    // ---- epilogue ----
    const float bscale = (wsA ? *wsA : 1.0f) + (wsB ? *wsB : 0.0f);
    #pragma unroll
    for (int i = 0; i < 4; ++i) {
        #pragma unroll
        for (int j = 0; j < 2; ++j) {
            #pragma unroll
            for (int r = 0; r < 8; ++r) {
                const int m = bm + waveM * 64 + i * 16 + r + hh * 8;   // C/D layout rows
                const int n = bn + waveN * 32 + j * 16 + lr;
                const float val = acc[i][j][r] + bias[n] * bscale;
                if (mode == 2) {
                    ((float*)Cout)[(size_t)m * NDIM + n] = val;
                } else {
                    const int b = m >> 10, s = m & 1023;
                    const int h = n >> 6,  d = n & 63;
                    const int idx = (mode == 0)
                        ? (((b * 12 + h) * 1024 + s) * 64 + d)     // [bh][s][d]
                        : (((b * 12 + h) * 64 + d) * 1024 + s);    // [bh][d][s]
                    ((unsigned short*)Cout)[idx] = f2bf(val);
                }
            }
        }
    }
}

// ---------------------------------------------------------------------------
// Flash attention: 4 waves/block, all on the same (b,h); each wave owns a
// 16-query tile.  K/V tiles for the 32-key step are staged once per block
// into LDS with async global->LDS copies and shared by all 4 waves.
// Q,K : bf16 [96][1024][64]   Vt : bf16 [96][64][1024]
// Ctx : f32  [8][1024][768]   (accumulate=1 -> Ctx += w*attn, else Ctx = w*attn)
// ---------------------------------------------------------------------------
__global__ __launch_bounds__(128)
void attn_flash_wmma(const unsigned short* __restrict__ Q,
                     const unsigned short* __restrict__ Kt,
                     const unsigned short* __restrict__ Vt,
                     const float* __restrict__ wptr,
                     float* __restrict__ Ctx, int accumulate)
{
    __shared__ unsigned short Ks[32 * 64];        // [key][d]   4KB, shared by block
    __shared__ unsigned short Vs[64 * 32];        // [d][key]   4KB, shared by block
    __shared__ unsigned short Plds[4 * 16 * 40];  // per-wave 16x32 P tile (+pad)

    const int t     = threadIdx.x;
    const int lane  = t & 31;
    const int wave  = t >> 5;
    const int hh    = lane >> 4;
    const int lr    = lane & 15;
    const int wg    = blockIdx.x * 4 + wave;   // 0..6143
    const int qtile = wg & 63;                 // blocks of 4 never straddle a head
    const int bh    = wg >> 6;                 // 0..95
    const int b     = bh / 12, h = bh % 12;
    const int qbase = qtile * 16;
    unsigned short* P = &Plds[wave * 16 * 40];

    // staging coordinates (block-cooperative)
    const unsigned short* Khead = Kt + (size_t)bh * 1024 * 64;  // contiguous rows
    const int vrow = t >> 1, vseg = t & 1;                      // V: 64 rows x 2 chunks

    // Q fragments (A-layout), d split into two K=32 chunks
    Frag aq[2];
    #pragma unroll
    for (int f = 0; f < 2; ++f) {
        const unsigned short* qp = &Q[(size_t)(bh * 1024 + qbase + lr) * 64 + f * 32 + hh * 8];
        aq[f].u4[0] = *(const uint4*)qp;
        aq[f].u4[1] = *(const uint4*)(qp + 16);
    }

    float m_run[8], l_run[8];
    #pragma unroll
    for (int r = 0; r < 8; ++r) { m_run[r] = -1e30f; l_run[r] = 0.0f; }
    v8f cacc[4] = {};

    for (int kb = 0; kb < 1024; kb += 32) {
        __syncthreads();   // previous iteration's K/V/P readers are done

        // ---- async-stage K tile (32x64, contiguous 4KB) and V^T tile ----
        {
            const unsigned short* gk = Khead + (size_t)kb * 64 + t * 16;
            async_copy16(gk,     &Ks[t * 16]);
            async_copy16(gk + 8, &Ks[t * 16 + 8]);
            const unsigned short* gv =
                Vt + (size_t)(bh * 64 + vrow) * 1024 + kb + vseg * 16;
            async_copy16(gv,     &Vs[vrow * 32 + vseg * 16]);
            async_copy16(gv + 8, &Vs[vrow * 32 + vseg * 16 + 8]);
        }
        async_wait0();
        __syncthreads();   // tiles visible to all waves

        // ---- scores: two 16x16 key tiles, K-dim = DH = 64 (2 WMMA each) ----
        v8f s0 = {}, s1 = {};
        #pragma unroll
        for (int f = 0; f < 2; ++f) {
            Frag bk0, bk1;
            const unsigned short* kp0 = &Ks[(0 + lr) * 64 + f * 32 + hh * 8];
            bk0.u4[0] = *(const uint4*)kp0;  bk0.u4[1] = *(const uint4*)(kp0 + 16);
            const unsigned short* kp1 = &Ks[(16 + lr) * 64 + f * 32 + hh * 8];
            bk1.u4[0] = *(const uint4*)kp1;  bk1.u4[1] = *(const uint4*)(kp1 + 16);
            s0 = wmma_bf16(aq[f], bk0, s0);
            s1 = wmma_bf16(aq[f], bk1, s1);
        }
        s0 *= 0.125f;   // 1/sqrt(64)
        s1 *= 0.125f;

        // ---- online softmax: row reductions across 16-lane halves ----
        #pragma unroll
        for (int r = 0; r < 8; ++r) {
            float mb = fmaxf(s0[r], s1[r]);
            mb = fmaxf(mb, __shfl_xor(mb, 1, 32));
            mb = fmaxf(mb, __shfl_xor(mb, 2, 32));
            mb = fmaxf(mb, __shfl_xor(mb, 4, 32));
            mb = fmaxf(mb, __shfl_xor(mb, 8, 32));
            const float newm  = fmaxf(m_run[r], mb);
            const float scale = __expf(m_run[r] - newm);
            const float p0 = __expf(s0[r] - newm);
            const float p1 = __expf(s1[r] - newm);
            float rs = p0 + p1;
            rs += __shfl_xor(rs, 1, 32);
            rs += __shfl_xor(rs, 2, 32);
            rs += __shfl_xor(rs, 4, 32);
            rs += __shfl_xor(rs, 8, 32);
            l_run[r] = l_run[r] * scale + rs;
            m_run[r] = newm;
            #pragma unroll
            for (int j = 0; j < 4; ++j) cacc[j][r] *= scale;
            // C-layout -> LDS (row r+8*hh, cols lr / 16+lr)
            P[(r + hh * 8) * 40 + lr]      = f2bf(p0);
            P[(r + hh * 8) * 40 + 16 + lr] = f2bf(p1);
        }
        __syncthreads();   // P visible

        // ---- P as A-fragment (16x32), V^T as B-fragments; ctx += P @ V ----
        Frag pf;
        pf.u4[0] = *(const uint4*)&P[lr * 40 + hh * 8];
        pf.u4[1] = *(const uint4*)&P[lr * 40 + 16 + hh * 8];
        #pragma unroll
        for (int j = 0; j < 4; ++j) {
            Frag bv;
            const unsigned short* vp = &Vs[(j * 16 + lr) * 32 + hh * 8];
            bv.u4[0] = *(const uint4*)vp;  bv.u4[1] = *(const uint4*)(vp + 16);
            cacc[j] = wmma_bf16(pf, bv, cacc[j]);
        }
    }

    // ---- normalize, blend-weight, write f32 context [b][s][h*64+d] ----
    const float w = *wptr;
    #pragma unroll
    for (int r = 0; r < 8; ++r) {
        const float inv = w / l_run[r];
        const int s = qbase + r + hh * 8;
        #pragma unroll
        for (int j = 0; j < 4; ++j) {
            const int col = h * 64 + j * 16 + lr;
            const size_t idx = (size_t)(b * 1024 + s) * 768 + col;
            const float val = cacc[j][r] * inv;
            if (accumulate) Ctx[idx] += val;
            else            Ctx[idx]  = val;
        }
    }
}

// ---------------------------------------------------------------------------
extern "C" void kernel_launch(void* const* d_in, const int* in_sizes, int n_in,
                              void* d_out, int out_size, void* d_ws, size_t ws_size,
                              hipStream_t stream)
{
    const float* hx  = (const float*)d_in[0];
    const float* hy  = (const float*)d_in[1];
    const float* Wq  = (const float*)d_in[2];  const float* bq  = (const float*)d_in[3];
    const float* Wk  = (const float*)d_in[4];  const float* bk  = (const float*)d_in[5];
    const float* Wv  = (const float*)d_in[6];  const float* bv  = (const float*)d_in[7];
    const float* Wo  = (const float*)d_in[8];  const float* bo  = (const float*)d_in[9];
    const float* Wqd = (const float*)d_in[10]; const float* bqd = (const float*)d_in[11];
    const float* Wkd = (const float*)d_in[12]; const float* bkd = (const float*)d_in[13];
    const float* Wvd = (const float*)d_in[14]; const float* bvd = (const float*)d_in[15];
    const float* Wod = (const float*)d_in[16]; const float* bod = (const float*)d_in[17];
    const float* w11 = (const float*)d_in[18]; const float* w12 = (const float*)d_in[19];
    const float* w21 = (const float*)d_in[20]; const float* w22 = (const float*)d_in[21];

    char* ws = (char*)d_ws;
    const size_t szQ   = (size_t)96 * 1024 * 64 * 2;       // bf16 head tensor
    const size_t szCtx = (size_t)8 * 1024 * 768 * 4;       // f32 context
    unsigned short* Qb  = (unsigned short*)(ws + 0 * szQ);
    unsigned short* Kb  = (unsigned short*)(ws + 1 * szQ);
    unsigned short* Vtb = (unsigned short*)(ws + 2 * szQ);
    unsigned short* Qd  = (unsigned short*)(ws + 3 * szQ);
    unsigned short* Kd  = (unsigned short*)(ws + 4 * szQ);
    unsigned short* Vtd = (unsigned short*)(ws + 5 * szQ);
    float* CtxX = (float*)(ws + 6 * szQ);
    float* CtxY = (float*)(ws + 6 * szQ + szCtx);

    dim3 gg(MDIM / 128, NDIM / 128), bg(256);
    // QKV projections (bias included; split-head / transposed-V epilogues)
    gemm_bf16wmma<<<gg, bg, 0, stream>>>(hx, Wq,  bq,  nullptr, nullptr, Qb,  0);
    gemm_bf16wmma<<<gg, bg, 0, stream>>>(hx, Wk,  bk,  nullptr, nullptr, Kb,  0);
    gemm_bf16wmma<<<gg, bg, 0, stream>>>(hx, Wv,  bv,  nullptr, nullptr, Vtb, 1);
    gemm_bf16wmma<<<gg, bg, 0, stream>>>(hy, Wqd, bqd, nullptr, nullptr, Qd,  0);
    gemm_bf16wmma<<<gg, bg, 0, stream>>>(hy, Wkd, bkd, nullptr, nullptr, Kd,  0);
    gemm_bf16wmma<<<gg, bg, 0, stream>>>(hy, Wvd, bvd, nullptr, nullptr, Vtd, 1);

    // 4 attentions -> 2 blended contexts (blend before the output projection)
    dim3 ga(1536), ba(128);
    attn_flash_wmma<<<ga, ba, 0, stream>>>(Qb, Kb, Vtb, w11, CtxX, 0); // self  x
    attn_flash_wmma<<<ga, ba, 0, stream>>>(Qb, Kd, Vtd, w12, CtxX, 1); // cross x
    attn_flash_wmma<<<ga, ba, 0, stream>>>(Qd, Kd, Vtd, w21, CtxY, 0); // self  y
    attn_flash_wmma<<<ga, ba, 0, stream>>>(Qd, Kb, Vtb, w22, CtxY, 1); // cross y

    // Output projections; bias scaled by (w11+w12) / (w21+w22)
    float* outX = (float*)d_out;
    float* outY = outX + (size_t)8 * 1024 * 768;
    gemm_bf16wmma<<<gg, bg, 0, stream>>>(CtxX, Wo,  bo,  w11, w12, outX, 2);
    gemm_bf16wmma<<<gg, bg, 0, stream>>>(CtxY, Wod, bod, w21, w22, outY, 2);
}
